// G3Median_20469814133060
// MI455X (gfx1250) — compile-verified, matrix operand order checked by reading, not run on hardware
//
#include <hip/hip_runtime.h>

typedef __attribute__((ext_vector_type(16))) __bf16 v16bf;
typedef __attribute__((ext_vector_type(8)))  __bf16 v8bf;
typedef __attribute__((ext_vector_type(8)))  float  v8f;

#define NN    15000
#define NE    150000
#define FEAT  75
#define FP    80      // padded feature stride (f32 rows)
#define TW    5
#define FOT   15
#define KPRE  160     // 2*75 -> padded to 5*32
#define NPRE  384     // 5*75 -> padded to 24*16
#define KPOST 992     // 975  -> padded to 31*32
#define KLIN  96      // 75   -> padded to 3*32
#define OCH   32
#define NGCN  64
#define EPSV  1e-5f
#define INV_LOG11 0.4170323914242463f

// ---------- helpers ----------
static __device__ __forceinline__ unsigned f2key(float f) {
  unsigned u = __float_as_uint(f);
  return (u & 0x80000000u) ? ~u : (u | 0x80000000u);
}
static __device__ __forceinline__ float key2f(unsigned k) {
  unsigned u = (k & 0x80000000u) ? (k & 0x7fffffffu) : ~k;
  return __uint_as_float(u);
}
static __device__ __forceinline__ v16bf ldB16(const __bf16* p) {
  v8bf lo = *(const v8bf*)p;
  v8bf hi = *(const v8bf*)(p + 16);
  v16bf r;
#pragma unroll
  for (int i = 0; i < 8; ++i) { r[i] = lo[i]; r[i + 8] = hi[i]; }
  return r;
}

// ---------- setup kernels ----------
__global__ void gather_kernel(const int* __restrict__ x, const float* __restrict__ emb,
                              float* __restrict__ h) {
  int idx = blockIdx.x * blockDim.x + threadIdx.x;
  if (idx >= NN * FP) return;
  int n = idx / FP, f = idx % FP;
  h[idx] = (f < FEAT) ? emb[x[n] * FEAT + f] : 0.f;
}

__global__ void zero_kernel(float* __restrict__ p, int n) {
  int i = blockIdx.x * blockDim.x + threadIdx.x;
  if (i < n) p[i] = 0.f;
}

__global__ void cnt_kernel(const int* __restrict__ dst, float* __restrict__ cnt) {
  int e = blockIdx.x * blockDim.x + threadIdx.x;
  if (e < NE) atomicAdd(&cnt[dst[e]], 1.0f);
}

__global__ void deg_kernel(const float* __restrict__ cnt, float* __restrict__ amp,
                           float* __restrict__ dinvg) {
  int n = blockIdx.x * blockDim.x + threadIdx.x;
  if (n >= NN) return;
  float c = cnt[n];
  float deg = fmaxf(c, 1.0f);
  amp[n] = logf(deg + 1.0f) * INV_LOG11;
  dinvg[n] = rsqrtf(c + 1.0f);
}

// ---------- weight prep (f32 -> padded transposed bf16) ----------
__global__ void prep_pre_kernel(const float* __restrict__ pw, const float* __restrict__ pb,
                                int l, __bf16* __restrict__ W, float* __restrict__ B) {
  int idx = blockIdx.x * blockDim.x + threadIdx.x;
  if (idx >= NPRE * KPRE) return;
  int n = idx / KPRE, k = idx % KPRE;
  float v = 0.f;
  if (n < TW * FEAT && k < 2 * FEAT) {
    int t = n / FEAT, g = n % FEAT;
    v = pw[((size_t)(l * TW + t) * (2 * FEAT) + k) * FEAT + g];
  }
  W[idx] = (__bf16)v;
  if (k == 0) {
    float b = 0.f;
    if (n < TW * FEAT) { int t = n / FEAT, g = n % FEAT; b = pb[(l * TW + t) * FEAT + g]; }
    B[n] = b;
  }
}

__global__ void prep_post_kernel(const float* __restrict__ pw, const float* __restrict__ pb,
                                 int l, __bf16* __restrict__ W, float* __restrict__ B) {
  int idx = blockIdx.x * blockDim.x + threadIdx.x;
  if (idx >= TW * 16 * KPOST) return;
  int t = idx / (16 * KPOST);
  int rem = idx % (16 * KPOST);
  int n = rem / KPOST, k = rem % KPOST;
  float v = 0.f;
  if (n < FOT && k < 975) v = pw[((size_t)(l * TW + t) * 975 + k) * FOT + n];
  W[idx] = (__bf16)v;
  if (k == 0) B[t * 16 + n] = (n < FOT) ? pb[(l * TW + t) * FOT + n] : 0.f;
}

__global__ void prep_lin_kernel(const float* __restrict__ lw, const float* __restrict__ lb,
                                int l, __bf16* __restrict__ W, float* __restrict__ B) {
  int idx = blockIdx.x * blockDim.x + threadIdx.x;
  if (idx >= FP * KLIN) return;
  int n = idx / KLIN, k = idx % KLIN;
  float v = 0.f;
  if (n < FEAT && k < FEAT) v = lw[((size_t)l * FEAT + k) * FEAT + n];
  W[idx] = (__bf16)v;
  if (k == 0) B[n] = (n < FEAT) ? lb[l * FEAT + n] : 0.f;
}

__global__ void prep_gcn_kernel(const float* __restrict__ mw, const float* __restrict__ mb,
                                const float* __restrict__ lw, const float* __restrict__ lb,
                                __bf16* __restrict__ W, float* __restrict__ B) {
  int idx = blockIdx.x * blockDim.x + threadIdx.x;
  if (idx >= NGCN * KLIN) return;
  int n = idx / KLIN, k = idx % KLIN;
  float v = 0.f;
  if (k < FEAT) v = (n < OCH) ? mw[k * OCH + n] : lw[k * OCH + (n - OCH)];
  W[idx] = (__bf16)v;
  if (k == 0) B[n] = (n < OCH) ? mb[n] : lb[n - OCH];
}

// ---------- aggregation buffer init: [node][tower][4 comps][FP] ----------
__global__ void agg_init_kernel(unsigned* __restrict__ A) {
  int idx = blockIdx.x * blockDim.x + threadIdx.x;
  if (idx >= NN * TW * 4 * FP) return;
  int comp = (idx / FP) & 3;
  A[idx] = (comp == 2) ? 0xFFFFFFFFu : 0u;  // sum=0, sumsq=0, minkey=max, maxkey=0
}

// ---------- edge message GEMM (WMMA bf16) + fused scatter aggregation ----------
__global__ void __launch_bounds__(128)
edge_msg_kernel(const float* __restrict__ h, const int* __restrict__ src,
                const int* __restrict__ dst, const __bf16* __restrict__ W,
                const float* __restrict__ bias, float* __restrict__ AGG) {
  const int wave = threadIdx.x >> 5;
  const int lane = threadIdx.x & 31;
  const int tile = blockIdx.x * 4 + wave;
  const int e0 = tile * 16;
  if (e0 >= NE) return;

  __shared__ float sh[4][16 * 164];
  float* my = sh[wave];

  // stage 16 edge rows of h_e = [x[dst], x[src]] (padded to 160) into LDS
  for (int r = 0; r < 16; ++r) {
    int e = e0 + r;
    bool ok = e < NE;
    int sd = ok ? dst[e] : 0;
    int ss = ok ? src[e] : 0;
    for (int f = lane; f < KPRE; f += 32) {
      float v = 0.f;
      if (ok) {
        if (f < FEAT) v = h[(size_t)sd * FP + f];
        else if (f < 2 * FEAT) v = h[(size_t)ss * FP + (f - FEAT)];
      }
      my[r * 164 + f] = v;
    }
  }

  const int row = lane & 15;
  const int kb = (lane >> 4) << 3;
  // cache A fragments for all 5 K-steps
  v16bf afrag[5];
#pragma unroll
  for (int kk = 0; kk < 5; ++kk) {
    const float* p = &my[row * 164 + kk * 32 + kb];
#pragma unroll
    for (int i = 0; i < 8; ++i) afrag[kk][i] = (__bf16)p[i];
#pragma unroll
    for (int i = 0; i < 8; ++i) afrag[kk][8 + i] = (__bf16)p[16 + i];
  }

  const int col = lane & 15;
  const int rbase = (lane >> 4) << 3;
  for (int nt = 0; nt < 24; ++nt) {
    int n = nt * 16 + col;
    float b = bias[n];
    v8f acc;
#pragma unroll
    for (int r = 0; r < 8; ++r) acc[r] = b;
#pragma unroll
    for (int kk = 0; kk < 5; ++kk) {
      v16bf bfrag = ldB16(&W[(size_t)n * KPRE + kk * 32 + kb]);
      acc = __builtin_amdgcn_wmma_f32_16x16x32_bf16(false, afrag[kk], false, bfrag,
                                                    (short)0, acc, false, false);
    }
    if (n < TW * FEAT) {
      int t = n / FEAT, g = n % FEAT;
#pragma unroll
      for (int r = 0; r < 8; ++r) {
        int e = e0 + rbase + r;
        if (e < NE) {
          float v = acc[r];
          int d = dst[e];
          float* base = AGG + (size_t)(d * TW + t) * 4 * FP;
          atomicAdd(base + g, v);
          atomicAdd(base + FP + g, v * v);
          unsigned key = f2key(v);
          atomicMin(reinterpret_cast<unsigned*>(base + 2 * FP + g), key);
          atomicMax(reinterpret_cast<unsigned*>(base + 3 * FP + g), key);
        }
      }
    }
  }
}

// ---------- finalize: sum/sumsq/minkey/maxkey -> mean/mn/mx/std (in place) ----------
__global__ void finalize_kernel(float* __restrict__ AGG, const float* __restrict__ cnt) {
  int idx = blockIdx.x * blockDim.x + threadIdx.x;
  if (idx >= NN * TW * FP) return;
  int f = idx % FP;
  if (f >= FEAT) return;
  int nt = idx / FP;
  int node = nt / TW;
  float* base = AGG + (size_t)nt * 4 * FP;
  float s = base[f];
  float sq = base[FP + f];
  unsigned mnk = __float_as_uint(base[2 * FP + f]);
  unsigned mxk = __float_as_uint(base[3 * FP + f]);
  float c = cnt[node];
  float deg = fmaxf(c, 1.0f);
  float mean = s / deg;
  float msq = sq / deg;
  float sd = sqrtf(fmaxf(msq - mean * mean, 0.f) + EPSV);
  bool has = c > 0.f;
  float mn = has ? key2f(mnk) : 0.f;
  float mx = has ? key2f(mxk) : 0.f;
  base[f] = mean;
  base[FP + f] = mn;
  base[2 * FP + f] = mx;
  base[3 * FP + f] = sd;
}

// ---------- post-MLP per (node-tile, tower): K=975 (padded 992) -> 15 ----------
__global__ void __launch_bounds__(128)
post_kernel(const float* __restrict__ h, const float* __restrict__ AGG,
            const float* __restrict__ amp_, const __bf16* __restrict__ W,
            const float* __restrict__ bias, float* __restrict__ P) {
  const int wave = threadIdx.x >> 5;
  const int lane = threadIdx.x & 31;
  int gw = blockIdx.x * 4 + wave;
  int mtile = gw / TW, t = gw % TW;
  if (mtile * 16 >= NN) return;
  const int row = lane & 15;
  const int kb = (lane >> 4) << 3;
  const int col = lane & 15;
  int node = mtile * 16 + row;
  bool nok = node < NN;
  int nc = nok ? node : 0;
  float ampv = amp_[nc];
  float iamp = 1.f / ampv;
  const float* hp = h + (size_t)nc * FP;
  const float* ap = AGG + (size_t)(nc * TW + t) * 4 * FP;
  float b = bias[t * 16 + col];
  v8f acc;
#pragma unroll
  for (int r = 0; r < 8; ++r) acc[r] = b;

  for (int kk = 0; kk < 31; ++kk) {
    v16bf afrag;
#pragma unroll
    for (int half = 0; half < 2; ++half) {
      int kbase = kk * 32 + kb + half * 16;
#pragma unroll
      for (int i = 0; i < 8; ++i) {
        int k = kbase + i;
        float v = 0.f;
        if (nok) {
          if (k < FEAT) {
            v = hp[k];
          } else if (k < 975) {
            unsigned c = (unsigned)(k - FEAT);
            unsigned s = c / 300u;
            unsigned rr = c - s * 300u;
            unsigned comp = rr / 75u;
            unsigned f = rr - comp * 75u;
            float scale = (s == 0u) ? 1.f : ((s == 1u) ? ampv : iamp);
            v = ap[comp * FP + f] * scale;
          }
        }
        afrag[half * 8 + i] = (__bf16)v;
      }
    }
    v16bf bfrag = ldB16(&W[((size_t)t * 16 + col) * KPOST + kk * 32 + kb]);
    acc = __builtin_amdgcn_wmma_f32_16x16x32_bf16(false, afrag, false, bfrag,
                                                  (short)0, acc, false, false);
  }
  int rbase = (lane >> 4) << 3;
#pragma unroll
  for (int r = 0; r < 8; ++r) {
    int nrow = mtile * 16 + rbase + r;
    if (nrow < NN && col < FOT)
      P[(size_t)nrow * KLIN + t * FOT + col] = acc[r];
  }
}

// ---------- lin GEMM: P(N x 75, stride 96) @ lin_w(75x75) ----------
__global__ void __launch_bounds__(128)
lin_kernel(const float* __restrict__ P, const __bf16* __restrict__ W,
           const float* __restrict__ bias, float* __restrict__ hX) {
  const int wave = threadIdx.x >> 5;
  const int lane = threadIdx.x & 31;
  int gw = blockIdx.x * 4 + wave;
  int mtile = gw / 5, ntile = gw % 5;
  if (mtile * 16 >= NN) return;
  const int row = lane & 15;
  const int kb = (lane >> 4) << 3;
  const int col = lane & 15;
  int node = mtile * 16 + row;
  bool nok = node < NN;
  const float* pp = P + (size_t)(nok ? node : 0) * KLIN;
  int n = ntile * 16 + col;
  float b = bias[n];
  v8f acc;
#pragma unroll
  for (int r = 0; r < 8; ++r) acc[r] = b;
#pragma unroll
  for (int kk = 0; kk < 3; ++kk) {
    v16bf afrag;
#pragma unroll
    for (int half = 0; half < 2; ++half) {
      int kbase = kk * 32 + kb + half * 16;
#pragma unroll
      for (int i = 0; i < 8; ++i) {
        int k = kbase + i;
        float v = (nok && k < FEAT) ? pp[k] : 0.f;
        afrag[half * 8 + i] = (__bf16)v;
      }
    }
    v16bf bfrag = ldB16(&W[(size_t)n * KLIN + kk * 32 + kb]);
    acc = __builtin_amdgcn_wmma_f32_16x16x32_bf16(false, afrag, false, bfrag,
                                                  (short)0, acc, false, false);
  }
  int rbase = (lane >> 4) << 3;
#pragma unroll
  for (int r = 0; r < 8; ++r) {
    int nrow = mtile * 16 + rbase + r;
    if (nrow < NN && n < FEAT) hX[(size_t)nrow * KLIN + n] = acc[r];
  }
}

// ---------- batch norm ----------
__global__ void bn_reduce_kernel(const float* __restrict__ hX, float* __restrict__ mu,
                                 float* __restrict__ var) {
  int f = blockIdx.x;  // 0..74
  __shared__ float s1[256], s2[256];
  float a = 0.f, b = 0.f;
  for (int n = threadIdx.x; n < NN; n += 256) {
    float v = hX[(size_t)n * KLIN + f];
    a += v; b += v * v;
  }
  s1[threadIdx.x] = a; s2[threadIdx.x] = b;
  __syncthreads();
  for (int s = 128; s > 0; s >>= 1) {
    if (threadIdx.x < s) { s1[threadIdx.x] += s1[threadIdx.x + s]; s2[threadIdx.x] += s2[threadIdx.x + s]; }
    __syncthreads();
  }
  if (threadIdx.x == 0) {
    float m = s1[0] / (float)NN;
    mu[f] = m;
    var[f] = s2[0] / (float)NN - m * m;
  }
}

__global__ void bn_apply_kernel(const float* __restrict__ hX, const float* __restrict__ mu,
                                const float* __restrict__ var, const float* __restrict__ gamma,
                                const float* __restrict__ beta, float* __restrict__ h) {
  int idx = blockIdx.x * blockDim.x + threadIdx.x;
  if (idx >= NN * FP) return;
  int n = idx / FP, f = idx % FP;
  float v = 0.f;
  if (f < FEAT) {
    float x = hX[(size_t)n * KLIN + f];
    v = (x - mu[f]) * rsqrtf(var[f] + EPSV) * gamma[f] + beta[f];
    v = fmaxf(v, 0.f);
  }
  h[idx] = v;
}

// ---------- GCN head: xw = h @ [Wmu | Wls]  (75 -> 64) ----------
__global__ void __launch_bounds__(128)
gcn_xw_kernel(const float* __restrict__ h, const __bf16* __restrict__ W,
              float* __restrict__ xw) {
  const int wave = threadIdx.x >> 5;
  const int lane = threadIdx.x & 31;
  int mtile = blockIdx.x * 4 + wave;
  if (mtile * 16 >= NN) return;
  const int row = lane & 15;
  const int kb = (lane >> 4) << 3;
  const int col = lane & 15;
  int node = mtile * 16 + row;
  bool nok = node < NN;
  const float* hp = h + (size_t)(nok ? node : 0) * FP;
  v16bf afrag[3];
#pragma unroll
  for (int kk = 0; kk < 3; ++kk) {
#pragma unroll
    for (int half = 0; half < 2; ++half) {
      int kbase = kk * 32 + kb + half * 16;
#pragma unroll
      for (int i = 0; i < 8; ++i) {
        int k = kbase + i;
        afrag[kk][half * 8 + i] = (__bf16)((nok && k < FEAT) ? hp[k] : 0.f);
      }
    }
  }
  int rbase = (lane >> 4) << 3;
  for (int nt = 0; nt < 4; ++nt) {
    int n = nt * 16 + col;
    v8f acc = {};
#pragma unroll
    for (int kk = 0; kk < 3; ++kk) {
      v16bf bfrag = ldB16(&W[(size_t)n * KLIN + kk * 32 + kb]);
      acc = __builtin_amdgcn_wmma_f32_16x16x32_bf16(false, afrag[kk], false, bfrag,
                                                    (short)0, acc, false, false);
    }
#pragma unroll
    for (int r = 0; r < 8; ++r) {
      int nrow = mtile * 16 + rbase + r;
      if (nrow < NN) xw[(size_t)nrow * NGCN + n] = acc[r];
    }
  }
}

__global__ void gcn_self_kernel(const float* __restrict__ xw, const float* __restrict__ dinvg,
                                const float* __restrict__ bias, float* __restrict__ out) {
  int idx = blockIdx.x * blockDim.x + threadIdx.x;
  if (idx >= NN * NGCN) return;
  int n = idx / NGCN, c = idx % NGCN;
  float di = dinvg[n];
  float v = xw[idx] * di * di + bias[c];
  if (c < OCH) out[(size_t)n * OCH + c] = v;
  else out[(size_t)NN * OCH + (size_t)n * OCH + (c - OCH)] = v;
}

__global__ void gcn_edge_kernel(const float* __restrict__ xw, const int* __restrict__ src,
                                const int* __restrict__ dst, const float* __restrict__ dinvg,
                                float* __restrict__ out) {
  int idx = blockIdx.x * blockDim.x + threadIdx.x;
  if (idx >= NE * NGCN) return;
  int e = idx / NGCN, c = idx % NGCN;
  int s = src[e], d = dst[e];
  float norm = dinvg[s] * dinvg[d];
  float v = xw[(size_t)s * NGCN + c] * norm;
  float* p = (c < OCH) ? (out + (size_t)d * OCH + c)
                       : (out + (size_t)NN * OCH + (size_t)d * OCH + (c - OCH));
  atomicAdd(p, v);
}

// ---------- launcher ----------
extern "C" void kernel_launch(void* const* d_in, const int* in_sizes, int n_in,
                              void* d_out, int out_size, void* d_ws, size_t ws_size,
                              hipStream_t stream) {
  const int*   x        = (const int*)d_in[0];
  const int*   ei       = (const int*)d_in[1];
  const float* node_emb = (const float*)d_in[2];
  const float* pre_w    = (const float*)d_in[3];
  const float* pre_b    = (const float*)d_in[4];
  const float* post_w   = (const float*)d_in[5];
  const float* post_b   = (const float*)d_in[6];
  const float* lin_w    = (const float*)d_in[7];
  const float* lin_b    = (const float*)d_in[8];
  const float* bn_g     = (const float*)d_in[9];
  const float* bn_b     = (const float*)d_in[10];
  const float* mu_w     = (const float*)d_in[11];
  const float* mu_b     = (const float*)d_in[12];
  const float* ls_w     = (const float*)d_in[13];
  const float* ls_b     = (const float*)d_in[14];
  (void)in_sizes; (void)n_in; (void)out_size; (void)ws_size;
  const int* srcp = ei;
  const int* dstp = ei + NE;
  float* out = (float*)d_out;

  char* ws = (char*)d_ws;
  size_t off = 0;
  auto take = [&](size_t bytes) -> char* {
    char* p = ws + off;
    off = (off + bytes + 255) & ~(size_t)255;
    return p;
  };
  float*  h     = (float*)take((size_t)NN * FP * 4);
  float*  P     = (float*)take((size_t)NN * KLIN * 4);
  float*  hX    = (float*)take((size_t)NN * KLIN * 4);
  float*  AGG   = (float*)take((size_t)NN * TW * 4 * FP * 4);
  float*  cnt   = (float*)take((size_t)NN * 4);
  float*  amp   = (float*)take((size_t)NN * 4);
  float*  dinvg = (float*)take((size_t)NN * 4);
  float*  xw    = (float*)take((size_t)NN * NGCN * 4);
  float*  bnmu  = (float*)take(FP * 4);
  float*  bnvar = (float*)take(FP * 4);
  __bf16* Wpre  = (__bf16*)take((size_t)NPRE * KPRE * 2);
  float*  bpre  = (float*)take(NPRE * 4);
  __bf16* Wpost = (__bf16*)take((size_t)TW * 16 * KPOST * 2);
  float*  bpost = (float*)take(TW * 16 * 4);
  __bf16* Wlin  = (__bf16*)take((size_t)FP * KLIN * 2);
  float*  blin  = (float*)take(FP * 4);
  __bf16* Wgcn  = (__bf16*)take((size_t)NGCN * KLIN * 2);
  float*  bgcn  = (float*)take(NGCN * 4);

  const int T = 256;
  auto cdiv = [](long long a, long long b) -> int { return (int)((a + b - 1) / b); };

  gather_kernel<<<cdiv((long long)NN * FP, T), T, 0, stream>>>(x, node_emb, h);
  zero_kernel<<<cdiv(NN, T), T, 0, stream>>>(cnt, NN);
  cnt_kernel<<<cdiv(NE, T), T, 0, stream>>>(dstp, cnt);
  deg_kernel<<<cdiv(NN, T), T, 0, stream>>>(cnt, amp, dinvg);
  prep_gcn_kernel<<<cdiv(NGCN * KLIN, T), T, 0, stream>>>(mu_w, mu_b, ls_w, ls_b, Wgcn, bgcn);

  const int edgeTiles = (NE + 15) / 16;            // 9375
  const int nodeTiles = (NN + 15) / 16;            // 938
  for (int l = 0; l < 4; ++l) {
    prep_pre_kernel<<<cdiv(NPRE * KPRE, T), T, 0, stream>>>(pre_w, pre_b, l, Wpre, bpre);
    prep_post_kernel<<<cdiv(TW * 16 * KPOST, T), T, 0, stream>>>(post_w, post_b, l, Wpost, bpost);
    prep_lin_kernel<<<cdiv(FP * KLIN, T), T, 0, stream>>>(lin_w, lin_b, l, Wlin, blin);

    agg_init_kernel<<<cdiv((long long)NN * TW * 4 * FP, T), T, 0, stream>>>((unsigned*)AGG);
    edge_msg_kernel<<<cdiv(edgeTiles, 4), 128, 0, stream>>>(h, srcp, dstp, Wpre, bpre, AGG);
    finalize_kernel<<<cdiv((long long)NN * TW * FP, T), T, 0, stream>>>(AGG, cnt);
    post_kernel<<<cdiv(nodeTiles * TW, 4), 128, 0, stream>>>(h, AGG, amp, Wpost, bpost, P);
    lin_kernel<<<cdiv(nodeTiles * 5, 4), 128, 0, stream>>>(P, Wlin, blin, hX);
    bn_reduce_kernel<<<FEAT, 256, 0, stream>>>(hX, bnmu, bnvar);
    bn_apply_kernel<<<cdiv((long long)NN * FP, T), T, 0, stream>>>(
        hX, bnmu, bnvar, bn_g + l * FEAT, bn_b + l * FEAT, h);
  }

  gcn_xw_kernel<<<cdiv(nodeTiles, 4), 128, 0, stream>>>(h, Wgcn, xw);
  gcn_self_kernel<<<cdiv((long long)NN * NGCN, T), T, 0, stream>>>(xw, dinvg, bgcn, out);
  gcn_edge_kernel<<<cdiv((long long)NE * NGCN, T), T, 0, stream>>>(xw, srcp, dstp, dinvg, out);
}
